// LPair_5720896438536
// MI455X (gfx1250) — matrix operand order repaired and checked by reading, not compiled, not run on hardware
//
#include <hip/hip_runtime.h>
#include <hip/hip_bf16.h>
#include <math.h>

typedef __attribute__((ext_vector_type(16))) _Float16 v16h;
typedef __attribute__((ext_vector_type(8)))  float    v8f;

#define NB 4
#define NN 512
#define DF 48
#define D2 96

// A-fragment (16x32 f16, M x K) per-lane K map: element e (0..15), half h = lane>>4
__device__ __forceinline__ int kmapA(int e, int h) {
    return e + 8 * h + ((e >= 8) ? 8 : 0);
}

// Branch-free exact-form GELU: 0.5*x*(1+erf(x/sqrt(2))) with erf via
// Abramowitz-Stegun 7.1.26 (|err| <= 1.5e-7, well below f16 GEMM noise).
// Straight-line VALU: 1 v_rcp, 1 v_exp, ~10 fma/mul, no EXEC divergence.
__device__ __forceinline__ float gelu_exact(float x) {
    const float z  = x * 0.70710678118654752440f;
    const float az = fabsf(z);
    const float t  = __builtin_amdgcn_rcpf(fmaf(az, 0.3275911f, 1.0f));
    float p = fmaf(1.061405429f, t, -1.453152027f);
    p = fmaf(p, t, 1.421413741f);
    p = fmaf(p, t, -0.284496736f);
    p = fmaf(p, t, 0.254829592f);
    p = p * t;
    const float e  = __expf(-az * az);
    float er = fmaf(-p, e, 1.0f);          // erf(|z|)
    er = copysignf(er, z);
    return 0.5f * x * (1.0f + er);
}

// ---------------------------------------------------------------------------
// Kernel 1: A[b,n,:] = x[b,n,:] @ W1[0:48,:] + b1 ; Bm[b,n,:] = x[b,n,:] @ W1[48:96,:]
// one block per row (b*N+n), 96 threads (one per output feature)
// ---------------------------------------------------------------------------
__global__ __launch_bounds__(96) void lpair_precompute(
    const float* __restrict__ x, const float* __restrict__ W1,
    const float* __restrict__ b1, float* __restrict__ A, float* __restrict__ Bm)
{
    __shared__ float xs[DF];
    const int row = blockIdx.x;           // b*N + n
    const int k   = threadIdx.x;          // 0..95
    if (k < DF) xs[k] = x[(size_t)row * DF + k];
    __syncthreads();
    float a = 0.f, bb = 0.f;
#pragma unroll
    for (int t = 0; t < DF; ++t) {
        float xv = xs[t];
        a  += xv * W1[t * D2 + k];
        bb += xv * W1[(DF + t) * D2 + k];
    }
    A [(size_t)row * D2 + k] = a + b1[k];
    Bm[(size_t)row * D2 + k] = bb;
}

// ---------------------------------------------------------------------------
// Kernel 2: per (b,i): online-softmax over lower-triangular j-tiles of 16,
// p tile computed with v_wmma_f32_16x16x32_f16 (M=16 pairs, K=96, N=48).
// One wave32 per (b,i); 2 waves per block, pairing rows r and N-1-r so every
// block does a constant amount of tile work (load balance).
// ---------------------------------------------------------------------------
__global__ __launch_bounds__(64) void lpair_main(
    const float* __restrict__ A, const float* __restrict__ Bm,
    const float* __restrict__ W2, const float* __restrict__ b2,
    float* __restrict__ out)
{
    const int lane = threadIdx.x & 31;
    const int wid  = threadIdx.x >> 5;
    const int b    = blockIdx.x >> 8;                  // 256 blocks per batch
    const int r    = blockIdx.x & 255;
    const int i    = (wid == 0) ? r : (NN - 1 - r);    // pair small+large rows
    const int c16  = lane & 15;              // M index for A-frag / N index for B,C frags
    const int h    = lane >> 4;              // half of wave

    // --- W2 B-fragments: w2f[c][n], c = K-chunk (32), n = N-tile (16) ---
    v16h w2f[3][3];
#pragma unroll
    for (int c = 0; c < 3; ++c)
#pragma unroll
        for (int n = 0; n < 3; ++n)
#pragma unroll
            for (int e = 0; e < 16; ++e) {
                int K = c * 32 + kmapA(e, h);
                w2f[c][n][e] = (_Float16)W2[K * DF + n * 16 + c16];
            }

    float b2v[3];
#pragma unroll
    for (int n = 0; n < 3; ++n) b2v[n] = b2[n * 16 + c16];

    // --- per-lane slice of A_i (fragment order) ---
    float av[3][16];
    const float* Arow = A + (size_t)(b * NN + i) * D2;
#pragma unroll
    for (int c = 0; c < 3; ++c)
#pragma unroll
        for (int e = 0; e < 16; ++e)
            av[c][e] = Arow[c * 32 + kmapA(e, h)];

    float m_run = -INFINITY, l_run = 0.f;
    float acc[3] = {0.f, 0.f, 0.f};

    const int ntiles = (i >> 4) + 1;
    for (int t = 0; t < ntiles; ++t) {
        const int j0 = t << 4;
        const float* Brow = Bm + (size_t)(b * NN + j0 + c16) * D2;

        // C tiles init with b2 broadcast (C layout: M = v + 8h, N = c16)
        v8f C[3];
#pragma unroll
        for (int n = 0; n < 3; ++n)
#pragma unroll
            for (int v = 0; v < 8; ++v) C[n][v] = b2v[n];

        // One K-chunk at a time: build H fragment, immediately consume it in
        // 3 WMMAs -> only 8 VGPRs of H live, GELU of chunk c+1 overlaps WMMA c.
#pragma unroll
        for (int c = 0; c < 3; ++c) {
            v16h hf;
#pragma unroll
            for (int e = 0; e < 16; ++e) {
                float v = av[c][e] + Brow[c * 32 + kmapA(e, h)];
                hf[e] = (_Float16)gelu_exact(v);
            }
#pragma unroll
            for (int n = 0; n < 3; ++n)
                C[n] = __builtin_amdgcn_wmma_f32_16x16x32_f16(
                           false, hf, false, w2f[c][n],
                           (short)0, C[n], false, false);
        }

        // row squared-norms: reduce over the 16 feature-lanes of each half
        float q[8];
#pragma unroll
        for (int v = 0; v < 8; ++v)
            q[v] = C[0][v] * C[0][v] + C[1][v] * C[1][v] + C[2][v] * C[2][v];
#pragma unroll
        for (int m = 1; m <= 8; m <<= 1)
#pragma unroll
            for (int v = 0; v < 8; ++v)
                q[v] += __shfl_xor(q[v], m, 32);

        float s[8];
        float tmax = -INFINITY;
#pragma unroll
        for (int v = 0; v < 8; ++v) {
            int j = j0 + v + 8 * h;
            s[v] = (j <= i) ? __builtin_sqrtf(q[v]) : -INFINITY;
            tmax = fmaxf(tmax, s[v]);
        }
        tmax = fmaxf(tmax, __shfl_xor(tmax, 16, 32));
        float m_new = fmaxf(m_run, tmax);

        float wv[8], sw = 0.f;
#pragma unroll
        for (int v = 0; v < 8; ++v) { wv[v] = __expf(s[v] - m_new); sw += wv[v]; }
        sw += __shfl_xor(sw, 16, 32);

        float scale = __expf(m_run - m_new);
        l_run = l_run * scale + sw;
#pragma unroll
        for (int n = 0; n < 3; ++n) {
            float a2 = 0.f;
#pragma unroll
            for (int v = 0; v < 8; ++v) a2 += wv[v] * C[n][v];
            acc[n] = acc[n] * scale + a2;
        }
        m_run = m_new;
    }

    const float inv = 1.0f / l_run;
#pragma unroll
    for (int n = 0; n < 3; ++n) {
        float a2 = acc[n] + __shfl_xor(acc[n], 16, 32);  // combine the two j-halves
        if (h == 0)
            out[(size_t)(b * NN + i) * DF + n * 16 + c16] = a2 * inv;
    }
}

// ---------------------------------------------------------------------------
extern "C" void kernel_launch(void* const* d_in, const int* in_sizes, int n_in,
                              void* d_out, int out_size, void* d_ws, size_t ws_size,
                              hipStream_t stream) {
    const float* x  = (const float*)d_in[0];   // [4,512,48]
    const float* W1 = (const float*)d_in[1];   // [96,96]
    const float* b1 = (const float*)d_in[2];   // [96]
    const float* W2 = (const float*)d_in[3];   // [96,48]
    const float* b2 = (const float*)d_in[4];   // [48]
    float* out = (float*)d_out;                // [4,512,48]

    float* A  = (float*)d_ws;                  // [4*512, 96]
    float* Bm = A + (size_t)NB * NN * D2;      // [4*512, 96]

    lpair_precompute<<<NB * NN, 96, 0, stream>>>(x, W1, b1, A, Bm);
    lpair_main<<<(NB * NN) / 2, 64, 0, stream>>>(A, Bm, W2, b2, out);
}
